// Sender_Receiver_35055523070745
// MI455X (gfx1250) — matrix operand order repaired and checked by reading, not compiled, not run on hardware
//
#include <hip/hip_runtime.h>
#include <hip/hip_bf16.h>
#include <math.h>

typedef _Float16 half_t;
typedef __attribute__((ext_vector_type(16))) _Float16 v16h;
typedef __attribute__((ext_vector_type(8)))  _Float16 v8h;
typedef __attribute__((ext_vector_type(8)))  float    v8f;

#define BROWS   16384
#define TASKW   16
#define FEATW   256
#define XCOLS   (TASKW + FEATW)   // 272
#define VOCAB   128
#define NH      512               // SH == RH == 512
#define TSTEPS  10
#define NDOUT   10
#define KC      (NH + TASKW)      // 528

// 1 = use gfx1250 GLOBAL_LOAD_ASYNC_TO_LDS_B128 (ASYNCcnt path) for tile staging
#define ASYNC_COPY 1

#define CAT16(lo,hi) __builtin_shufflevector(lo,hi,0,1,2,3,4,5,6,7,8,9,10,11,12,13,14,15)

static __device__ __forceinline__ float sigm_(float x) { return 1.0f / (1.0f + __expf(-x)); }

// Stage one 16B chunk global -> LDS.
static __device__ __forceinline__ void stage16(half_t* l, const half_t* g) {
#if ASYNC_COPY
    // LDS byte offset = low 32 bits of the generic pointer (addrspacecast truncation).
    asm volatile("global_load_async_to_lds_b128 %0, %1, off"
                 :: "v"((unsigned)(uintptr_t)l), "v"(g)
                 : "memory");
#else
    *(int4*)l = *(const int4*)g;
#endif
}
static __device__ __forceinline__ void stage_wait() {
#if ASYNC_COPY
    asm volatile("s_wait_asynccnt 0x0" ::: "memory");
#endif
}

// Load one A-fragment (16x32 f16): lane-group g holds K in [8g,8g+8) U [16+8g,16+8g+8)
static __device__ __forceinline__ v16h frag_a(const half_t* row, int g) {
    v8h lo = *(const v8h*)(row + g * 8);
    v8h hi = *(const v8h*)(row + 16 + g * 8);
    return CAT16(lo, hi);
}
// Load one B-fragment (32x16 f16): lane-group g holds K in [8g,8g+8) U [16+8g,16+8g+8)
static __device__ __forceinline__ v16h frag_b(const half_t* row, int g) {
    v8h lo = *(const v8h*)(row + g * 8);
    v8h hi = *(const v8h*)(row + 16 + g * 8);
    return CAT16(lo, hi);
}

// ---------------------------------------------------------------------------
// small helpers
// ---------------------------------------------------------------------------
__global__ void cvt_f32_f16_kernel(const float* __restrict__ src, half_t* __restrict__ dst, int n) {
    int i = blockIdx.x * blockDim.x + threadIdx.x;
    if (i < n) dst[i] = (half_t)src[i];
}

__global__ void feats_f16_kernel(const float* __restrict__ x, half_t* __restrict__ out) {
    int i = blockIdx.x * blockDim.x + threadIdx.x;   // i < BROWS*FEATW
    if (i >= BROWS * FEATW) return;
    int m = i >> 8, c = i & 255;
    out[i] = (half_t)x[(size_t)m * XCOLS + TASKW + c];
}

__global__ void zero_h_kernel(float* __restrict__ h32, half_t* __restrict__ h16) {
    int i = blockIdx.x * blockDim.x + threadIdx.x;
    if (i < BROWS * NH) { h32[i] = 0.0f; h16[i] = (half_t)0.0f; }
}

// ---------------------------------------------------------------------------
// h0 = tanh(feats @ W_in^T + b_in) : M=16384 N=512 K=256  (BM=128 BN=64 BK=32)
// ---------------------------------------------------------------------------
__global__ __launch_bounds__(256)
void h0_kernel(const half_t* __restrict__ A16, const half_t* __restrict__ W16,
               const float* __restrict__ bias, float* __restrict__ H, half_t* __restrict__ H16) {
    constexpr int BM = 128, BN = 64, BK = 32, LDA = 40, K = FEATW;
    __shared__ half_t As[BM * LDA];
    __shared__ half_t Bs[BN * LDA];
    const int tid = threadIdx.x, wave = tid >> 5, lane = tid & 31;
    const int wm = wave & 3, wn = wave >> 2;          // 4 x 2 waves
    const int m0 = blockIdx.y * BM, n0 = blockIdx.x * BN;
    const int g = lane >> 4, rc = lane & 15;
    const int cr = tid >> 2, cc = tid & 3;            // staging: row-chunk, 16B column

    v8f acc[2][2] = {};
    for (int k0 = 0; k0 < K; k0 += BK) {
#pragma unroll
        for (int q = 0; q < 2; q++) {                 // A: 128 rows x 4 chunks
            int r = cr + q * 64;
            stage16(&As[r * LDA + cc * 8], &A16[(size_t)(m0 + r) * K + k0 + cc * 8]);
        }
        stage16(&Bs[cr * LDA + cc * 8], &W16[(size_t)(n0 + cr) * K + k0 + cc * 8]);
        stage_wait();
        __syncthreads();
        v16h a[2], b[2];
#pragma unroll
        for (int i = 0; i < 2; i++) a[i] = frag_a(&As[(wm * 32 + i * 16 + rc) * LDA], g);
#pragma unroll
        for (int j = 0; j < 2; j++) b[j] = frag_b(&Bs[(wn * 32 + j * 16 + rc) * LDA], g);
#pragma unroll
        for (int i = 0; i < 2; i++)
#pragma unroll
            for (int j = 0; j < 2; j++)
                acc[i][j] = __builtin_amdgcn_wmma_f32_16x16x32_f16(false, a[i], false, b[j],
                                                                   (short)0, acc[i][j], false, false);
        __syncthreads();
    }
#pragma unroll
    for (int i = 0; i < 2; i++)
#pragma unroll
        for (int j = 0; j < 2; j++)
#pragma unroll
            for (int v = 0; v < 8; v++) {
                int m = m0 + wm * 32 + i * 16 + v + ((lane >> 4) << 3);
                int n = n0 + wn * 32 + j * 16 + (lane & 15);
                float h = tanhf(acc[i][j][v] + bias[n]);
                H[(size_t)m * NH + n] = h;
                H16[(size_t)m * NH + n] = (half_t)h;
            }
}

// ---------------------------------------------------------------------------
// Fused GRU step: gh = h @ Whh^T (3 gates, WMMA), gi = Wih[:,word] gather,
// then full elementwise GRU update. BM=128 BN=64 BK=32, 3 gate tiles/WG.
// phase 0 = sender (input = onehot(word[t-1]), zero at t=0)
// phase 1 = receiver (input = onehot(word[t]) if all words[0..t-1]!=0 else 0)
// ---------------------------------------------------------------------------
__global__ __launch_bounds__(256)
void gru_step_kernel(const half_t* __restrict__ A16, const float* __restrict__ Hold,
                     const half_t* __restrict__ Whh16, const float* __restrict__ Wih,
                     const float* __restrict__ bih, const float* __restrict__ bhh,
                     const int* __restrict__ words, int t, int phase,
                     float* __restrict__ Hnew, half_t* __restrict__ Hnew16) {
    constexpr int BM = 128, BN = 64, BK = 32, LDA = 40, K = NH;
    __shared__ half_t As[BM * LDA];
    __shared__ half_t Bs[3 * BN * LDA];
    const int tid = threadIdx.x, wave = tid >> 5, lane = tid & 31;
    const int wm = wave & 3, wn = wave >> 2;          // 4 x 2 waves
    const int m0 = blockIdx.y * BM, n0 = blockIdx.x * BN;
    const int gsel = lane >> 4, rc = lane & 15;
    const int cr = tid >> 2, cc = tid & 3;

    v8f acc[3][2][2] = {};
    for (int k0 = 0; k0 < K; k0 += BK) {
#pragma unroll
        for (int q = 0; q < 2; q++) {                 // A tile: 128 rows x 4 chunks
            int r = cr + q * 64;
            stage16(&As[r * LDA + cc * 8], &A16[(size_t)(m0 + r) * K + k0 + cc * 8]);
        }
#pragma unroll
        for (int q = 0; q < 3; q++) {                 // B tiles: 3 gates x 64 rows x 4 chunks
            stage16(&Bs[(q * BN + cr) * LDA + cc * 8],
                    &Whh16[(size_t)(q * NH + n0 + cr) * K + k0 + cc * 8]);
        }
        stage_wait();
        __syncthreads();
        v16h a[2];
#pragma unroll
        for (int i = 0; i < 2; i++) a[i] = frag_a(&As[(wm * 32 + i * 16 + rc) * LDA], gsel);
#pragma unroll
        for (int gg = 0; gg < 3; gg++)
#pragma unroll
            for (int j = 0; j < 2; j++) {
                v16h b = frag_b(&Bs[(gg * BN + wn * 32 + j * 16 + rc) * LDA], gsel);
#pragma unroll
                for (int i = 0; i < 2; i++)
                    acc[gg][i][j] = __builtin_amdgcn_wmma_f32_16x16x32_f16(
                        false, a[i], false, b, (short)0, acc[gg][i][j], false, false);
            }
        __syncthreads();
    }
    // epilogue: fused GRU update
#pragma unroll
    for (int i = 0; i < 2; i++)
#pragma unroll
        for (int v = 0; v < 8; v++) {
            int m = m0 + wm * 32 + i * 16 + v + ((lane >> 4) << 3);
            int widx = -1;   // -1 => zero input vector
            if (phase == 0) {
                if (t > 0) widx = words[(t - 1) * BROWS + m];
            } else {
                bool alive = true;
                for (int s = 0; s < t; s++) alive = alive && (words[s * BROWS + m] != 0);
                if (alive) widx = words[t * BROWS + m];
            }
#pragma unroll
            for (int j = 0; j < 2; j++) {
                int n = n0 + wn * 32 + j * 16 + (lane & 15);
                float ghr = acc[0][i][j][v] + bhh[n];
                float ghz = acc[1][i][j][v] + bhh[NH + n];
                float ghn = acc[2][i][j][v] + bhh[2 * NH + n];
                float gir = bih[n], giz = bih[NH + n], gin = bih[2 * NH + n];
                if (widx >= 0) {
                    gir += Wih[(size_t)n * VOCAB + widx];
                    giz += Wih[(size_t)(NH + n) * VOCAB + widx];
                    gin += Wih[(size_t)(2 * NH + n) * VOCAB + widx];
                }
                float r = sigm_(gir + ghr);
                float z = sigm_(giz + ghz);
                float nn = tanhf(gin + r * ghn);
                float hold = Hold[(size_t)m * NH + n];
                float hnew = (1.0f - z) * nn + z * hold;
                Hnew[(size_t)m * NH + n] = hnew;
                Hnew16[(size_t)m * NH + n] = (half_t)hnew;
            }
        }
}

// ---------------------------------------------------------------------------
// logits = h @ W_sp^T + b_sp + gumbel[t]; word[t] = argmax over VOCAB=128
// BM=64 BN=128 BK=32; logits tile staged in LDS for the row-argmax.
// ---------------------------------------------------------------------------
__global__ __launch_bounds__(256)
void logits_kernel(const half_t* __restrict__ A16, const half_t* __restrict__ Wsp16,
                   const float* __restrict__ bsp, const float* __restrict__ gumbel,
                   int* __restrict__ words, int t) {
    constexpr int BM = 64, BN = 128, BK = 32, LDA = 40, K = NH;
    __shared__ half_t As[BM * LDA];
    __shared__ half_t Bs[BN * LDA];
    __shared__ float  Lg[BM][BN + 4];
    const int tid = threadIdx.x, wave = tid >> 5, lane = tid & 31;
    const int wm = wave & 1, wn = wave >> 1;          // 2 x 4 waves
    const int m0 = blockIdx.y * BM;
    const int g = lane >> 4, rc = lane & 15;
    const int cr = tid >> 2, cc = tid & 3;

    v8f acc[2][2] = {};
    for (int k0 = 0; k0 < K; k0 += BK) {
        stage16(&As[cr * LDA + cc * 8], &A16[(size_t)(m0 + cr) * K + k0 + cc * 8]);
#pragma unroll
        for (int q = 0; q < 2; q++) {                 // B: 128 rows x 4 chunks
            int r = cr + q * 64;
            stage16(&Bs[r * LDA + cc * 8], &Wsp16[(size_t)r * K + k0 + cc * 8]);
        }
        stage_wait();
        __syncthreads();
        v16h a[2], b[2];
#pragma unroll
        for (int i = 0; i < 2; i++) a[i] = frag_a(&As[(wm * 32 + i * 16 + rc) * LDA], g);
#pragma unroll
        for (int j = 0; j < 2; j++) b[j] = frag_b(&Bs[(wn * 32 + j * 16 + rc) * LDA], g);
#pragma unroll
        for (int i = 0; i < 2; i++)
#pragma unroll
            for (int j = 0; j < 2; j++)
                acc[i][j] = __builtin_amdgcn_wmma_f32_16x16x32_f16(false, a[i], false, b[j],
                                                                   (short)0, acc[i][j], false, false);
        __syncthreads();
    }
#pragma unroll
    for (int i = 0; i < 2; i++)
#pragma unroll
        for (int j = 0; j < 2; j++)
#pragma unroll
            for (int v = 0; v < 8; v++) {
                int ml = wm * 32 + i * 16 + v + ((lane >> 4) << 3);
                int nl = wn * 32 + j * 16 + (lane & 15);
                Lg[ml][nl] = acc[i][j][v] + bsp[nl];
            }
    __syncthreads();
    if (tid < BM) {
        int m = m0 + tid;
        const float* gm = gumbel + ((size_t)t * BROWS + m) * VOCAB;
        float best = -1e30f; int bi = 0;
        for (int n = 0; n < VOCAB; n++) {
            float val = Lg[tid][n] + gm[n];
            if (val > best) { best = val; bi = n; }
        }
        words[t * BROWS + m] = bi;
    }
}

// ---------------------------------------------------------------------------
// score[m,d] = [h_n, task] @ W_c^T + b_c    (K=528, N=10 — scalar dot per thread)
// ---------------------------------------------------------------------------
__global__ void score_kernel(const float* __restrict__ h, const float* __restrict__ x,
                             const float* __restrict__ Wc, const float* __restrict__ bc,
                             float* __restrict__ out) {
    int i = blockIdx.x * blockDim.x + threadIdx.x;
    if (i >= BROWS * NDOUT) return;
    int m = i / NDOUT, d = i % NDOUT;
    const float* w = Wc + (size_t)d * KC;
    float s = bc[d];
    const float* hr = h + (size_t)m * NH;
    for (int k = 0; k < NH; k++) s += hr[k] * w[k];
    const float* xr = x + (size_t)m * XCOLS;
    for (int j = 0; j < TASKW; j++) s += xr[j] * w[NH + j];
    out[i] = s;
}

// ---------------------------------------------------------------------------
extern "C" void kernel_launch(void* const* d_in, const int* in_sizes, int n_in,
                              void* d_out, int out_size, void* d_ws, size_t ws_size,
                              hipStream_t stream) {
    const float* x     = (const float*)d_in[0];
    const float* gum   = (const float*)d_in[1];
    const float* W_in  = (const float*)d_in[2];
    const float* b_in  = (const float*)d_in[3];
    const float* Wih_s = (const float*)d_in[4];
    const float* Whh_s = (const float*)d_in[5];
    const float* bih_s = (const float*)d_in[6];
    const float* bhh_s = (const float*)d_in[7];
    const float* W_sp  = (const float*)d_in[8];
    const float* b_sp  = (const float*)d_in[9];
    const float* Wih_r = (const float*)d_in[10];
    const float* Whh_r = (const float*)d_in[11];
    const float* bih_r = (const float*)d_in[12];
    const float* bhh_r = (const float*)d_in[13];
    const float* W_c   = (const float*)d_in[14];
    const float* b_c   = (const float*)d_in[15];
    float* out = (float*)d_out;

    char* p = (char*)d_ws;
    auto carve = [&](size_t bytes) { void* r = (void*)p; p += (bytes + 255) & ~(size_t)255; return r; };
    float*  h32a    = (float*) carve((size_t)BROWS * NH * 4);
    float*  h32b    = (float*) carve((size_t)BROWS * NH * 4);
    half_t* h16a    = (half_t*)carve((size_t)BROWS * NH * 2);
    half_t* h16b    = (half_t*)carve((size_t)BROWS * NH * 2);
    half_t* feats16 = (half_t*)carve((size_t)BROWS * FEATW * 2);
    half_t* Win16   = (half_t*)carve((size_t)NH * FEATW * 2);
    half_t* Whs16   = (half_t*)carve((size_t)3 * NH * NH * 2);
    half_t* Whr16   = (half_t*)carve((size_t)3 * NH * NH * 2);
    half_t* Wsp16   = (half_t*)carve((size_t)VOCAB * NH * 2);
    int*    words   = (int*)   carve((size_t)TSTEPS * BROWS * 4);

    const int TB = 256;
    cvt_f32_f16_kernel<<<(NH * FEATW + TB - 1) / TB, TB, 0, stream>>>(W_in, Win16, NH * FEATW);
    cvt_f32_f16_kernel<<<(3 * NH * NH + TB - 1) / TB, TB, 0, stream>>>(Whh_s, Whs16, 3 * NH * NH);
    cvt_f32_f16_kernel<<<(3 * NH * NH + TB - 1) / TB, TB, 0, stream>>>(Whh_r, Whr16, 3 * NH * NH);
    cvt_f32_f16_kernel<<<(VOCAB * NH + TB - 1) / TB, TB, 0, stream>>>(W_sp, Wsp16, VOCAB * NH);
    feats_f16_kernel<<<(BROWS * FEATW + TB - 1) / TB, TB, 0, stream>>>(x, feats16);

    // h0 = tanh(feats @ W_in^T + b_in)
    h0_kernel<<<dim3(NH / 64, BROWS / 128), 256, 0, stream>>>(feats16, Win16, b_in, h32a, h16a);

    float*  hc32 = h32a, *hn32 = h32b;
    half_t* hc16 = h16a, *hn16 = h16b;

    // sender: 10 GRU steps, each followed by logits+argmax
    for (int t = 0; t < TSTEPS; t++) {
        gru_step_kernel<<<dim3(NH / 64, BROWS / 128), 256, 0, stream>>>(
            hc16, hc32, Whs16, Wih_s, bih_s, bhh_s, words, t, 0, hn32, hn16);
        { float* tf = hc32; hc32 = hn32; hn32 = tf; half_t* th = hc16; hc16 = hn16; hn16 = th; }
        logits_kernel<<<dim3(1, BROWS / 64), 256, 0, stream>>>(hc16, Wsp16, b_sp, gum, words, t);
    }

    // receiver: restart from h = 0
    zero_h_kernel<<<(BROWS * NH + TB - 1) / TB, TB, 0, stream>>>(hc32, hc16);
    for (int t = 0; t < TSTEPS; t++) {
        gru_step_kernel<<<dim3(NH / 64, BROWS / 128), 256, 0, stream>>>(
            hc16, hc32, Whr16, Wih_r, bih_r, bhh_r, words, t, 1, hn32, hn16);
        { float* tf = hc32; hc32 = hn32; hn32 = tf; half_t* th = hc16; hc16 = hn16; hn16 = th; }
    }

    // classifier
    score_kernel<<<(BROWS * NDOUT + TB - 1) / TB, TB, 0, stream>>>(hc32, x, W_c, b_c, out);
}
